// EncoderNetwork_38079180046560
// MI455X (gfx1250) — matrix operand order repaired and checked by reading, not compiled, not run on hardware
//
#include <hip/hip_runtime.h>
#include <hip/hip_bf16.h>
#include <math.h>
#include <stdint.h>

#define BATCH 4096
#define BM 128
#define BN 128
#define BK 32
#define LDT 40   // padded bf16 elements per LDS tile row (80B: conflict-free b128 frag reads)

typedef __bf16 bf16;
typedef __attribute__((ext_vector_type(16))) __bf16 bf16x16;
typedef __attribute__((ext_vector_type(8)))  __bf16 bf16x8;
typedef __attribute__((ext_vector_type(8)))  float  f32x8;

// ---------------------------------------------------------------------------
// CDNA5 async global->LDS copy (ASYNCcnt-tracked), guarded so the file also
// compiles on toolchains without the builtin (and in the host pass).
// Builtin signature (from hipcc diagnostic): param0 is a pointer to
// 'int __attribute__((vector_size(16)))' (the b128 payload type).
// ---------------------------------------------------------------------------
#if defined(__AMDGCN__) && __has_builtin(__builtin_amdgcn_global_load_async_to_lds_b128)
#define USE_ASYNC_LDS 1
#endif

#ifdef USE_ASYNC_LDS
typedef int v4i_vs __attribute__((vector_size(16)));
typedef __attribute__((address_space(1))) v4i_vs gbl_v4i;
typedef __attribute__((address_space(3))) v4i_vs lds_v4i;

__device__ __forceinline__ void async_cp_b128(const void* g, void* l) {
  // generic global addr == AS1 addr; generic LDS addr[31:0] == LDS byte offset
  __builtin_amdgcn_global_load_async_to_lds_b128(
      (gbl_v4i*)(uintptr_t)g,
      (lds_v4i*)(uint32_t)(uintptr_t)l, 0, 0);
}

template <int N> __device__ __forceinline__ void wait_async() {
#if __has_builtin(__builtin_amdgcn_s_wait_asynccnt)
  __builtin_amdgcn_s_wait_asynccnt(N);
#else
  asm volatile("s_wait_asynccnt %0" ::"n"(N));
#endif
}
#endif

// ---------------------------------------------------------------------------
// concat(y, u) -> x [BATCH, 512] f32
// ---------------------------------------------------------------------------
__global__ void kan_concat(const float* __restrict__ y, const float* __restrict__ u,
                           float* __restrict__ x) {
  int t = blockIdx.x * 256 + threadIdx.x;
  if (t >= BATCH * 512) return;
  int b = t >> 9;
  int i = t & 511;
  x[t] = (i < 256) ? y[b * 256 + i] : u[b * 256 + (i - 256)];
}

// ---------------------------------------------------------------------------
// Phi pack: per (batch, input) element compute [silu(x), B0..B7(x)] -> bf16
// Uniform pykan grid: h = 2/5, knots t_j = -1 + (j-3)*h, j = 0..11.
// Cox-de Boor with constant denominators d*h.
// Phi is [BATCH, Kdim] row-major, Kdim = 9*in_dim.
// ---------------------------------------------------------------------------
__global__ void kan_phi_pack(const float* __restrict__ x, int in_dim, int Kdim,
                             bf16* __restrict__ phi) {
  int t = blockIdx.x * 256 + threadIdx.x;
  if (t >= BATCH * in_dim) return;
  int b = t / in_dim;
  int i = t - b * in_dim;
  float xv = x[t];

  float s = xv / (1.0f + __expf(-xv));   // silu

  const float h = 0.4f;
  float bb[11];
#pragma unroll
  for (int j = 0; j < 11; ++j) {
    float tj = -1.0f + (float)(j - 3) * h;
    bb[j] = (xv >= tj && xv < tj + h) ? 1.0f : 0.0f;   // degree 0
  }
#pragma unroll
  for (int d = 1; d <= 3; ++d) {
    float inv = 1.0f / ((float)d * h);
#pragma unroll
    for (int j = 0; j < 10; ++j) {
      if (j < 11 - d) {
        float tj   = -1.0f + (float)(j - 3) * h;
        float tend = tj + (float)(d + 1) * h;
        bb[j] = (xv - tj) * inv * bb[j] + (tend - xv) * inv * bb[j + 1];
      }
    }
  }

  bf16* dst = phi + (size_t)b * Kdim + (size_t)i * 9;
  dst[0] = (bf16)s;
#pragma unroll
  for (int g = 0; g < 8; ++g) dst[1 + g] = (bf16)bb[g];
}

// ---------------------------------------------------------------------------
// Weight pack: Wt [out, Kdim] bf16 row-major (N-major so the GEMM is NT and
// B tiles load identically to A tiles).
// Wt[j][9i] = sb[i][j];  Wt[j][9i+1+g] = ss[i][j]*coef[i][j][g]
// ---------------------------------------------------------------------------
__global__ void kan_pack_w(const float* __restrict__ coef, const float* __restrict__ sb,
                           const float* __restrict__ ss, int in_dim, int out_dim,
                           int Kdim, bf16* __restrict__ wt) {
  int t = blockIdx.x * 256 + threadIdx.x;
  if (t >= in_dim * out_dim) return;
  int j = t / in_dim;
  int i = t - j * in_dim;
  size_t ij = (size_t)i * out_dim + j;
  float sbv = sb[ij];
  float ssv = ss[ij];
  const float* cf = coef + ij * 8;
  bf16* dst = wt + (size_t)j * Kdim + (size_t)i * 9;
  dst[0] = (bf16)sbv;
#pragma unroll
  for (int g = 0; g < 8; ++g) dst[1 + g] = (bf16)(ssv * cf[g]);
}

// ---------------------------------------------------------------------------
// NT GEMM: C[M,N] = A[M,K] * B[N,K]^T, bf16 in, f32 out, WMMA 16x16x32 bf16.
// Block tile 128x128, K-step 32, 256 threads = 8 waves (wave32), wave grid
// 4(m) x 2(n): each wave owns 32x64 = 2x4 WMMA tiles.
// Double-buffered LDS with CDNA5 async global->LDS copies: tile i+1 streams
// into the other buffer (ASYNCcnt) while tile i's WMMAs execute.
// ---------------------------------------------------------------------------
__global__ __launch_bounds__(256) void kan_gemm(
    const bf16* __restrict__ A, const bf16* __restrict__ B,
    float* __restrict__ C, int M, int N, int K) {
  __shared__ alignas(16) bf16 As[2][BM * LDT];
  __shared__ alignas(16) bf16 Bs[2][BN * LDT];

  const int tid  = threadIdx.x;
  const int wave = tid >> 5;
  const int lane = tid & 31;
  const int wm   = wave & 3;   // 0..3 -> 32 rows each
  const int wn   = wave >> 2;  // 0..1 -> 64 cols each

  const int block_m = blockIdx.x * BM;
  const int block_n = blockIdx.y * BN;

  const int lhalf = lane & 15;
  const int ksel  = lane >> 4;   // 0: lanes 0-15, 1: lanes 16-31

  // global->LDS mapping: 64 row-groups x 4 chunks of 8 halves (16B)
  const int ldrow = tid >> 2;
  const int ldcol = (tid & 3) * 8;

  const bf16*  gA      = A + (size_t)(block_m + ldrow) * K + ldcol;
  const bf16*  gB      = B + (size_t)(block_n + ldrow) * K + ldcol;
  const size_t rowskip = (size_t)64 * K;

  f32x8 acc[2][4] = {};

  auto load_tile = [&](int p, int kk) {
    bf16* la = &As[p][ldrow * LDT + ldcol];
    bf16* lb = &Bs[p][ldrow * LDT + ldcol];
#ifdef USE_ASYNC_LDS
    async_cp_b128(gA + kk,           la);
    async_cp_b128(gA + kk + rowskip, la + 64 * LDT);
    async_cp_b128(gB + kk,           lb);
    async_cp_b128(gB + kk + rowskip, lb + 64 * LDT);
#else
    *(bf16x8*)la             = *(const bf16x8*)(gA + kk);
    *(bf16x8*)(la + 64*LDT)  = *(const bf16x8*)(gA + kk + rowskip);
    *(bf16x8*)lb             = *(const bf16x8*)(gB + kk);
    *(bf16x8*)(lb + 64*LDT)  = *(const bf16x8*)(gB + kk + rowskip);
#endif
  };

  auto compute_tile = [&](int p) {
    // A fragment: lane<16 holds K 0-7 / 16-23, lane>=16 holds K 8-15 / 24-31
    bf16x16 afrag[2];
#pragma unroll
    for (int mi = 0; mi < 2; ++mi) {
      const bf16* q = &As[p][(wm * 32 + mi * 16 + lhalf) * LDT + ksel * 8];
      bf16x8 lo = *(const bf16x8*)q;
      bf16x8 hi = *(const bf16x8*)(q + 16);
      afrag[mi] = __builtin_shufflevector(lo, hi, 0, 1, 2, 3, 4, 5, 6, 7,
                                                  8, 9, 10, 11, 12, 13, 14, 15);
    }
    // B fragment: lane<16 holds K 0-7 / 16-23, lane>=16 holds K 8-15 / 24-31
    bf16x16 bfrag[4];
#pragma unroll
    for (int ni = 0; ni < 4; ++ni) {
      const bf16* q = &Bs[p][(wn * 64 + ni * 16 + lhalf) * LDT + ksel * 8];
      bf16x8 lo = *(const bf16x8*)q;
      bf16x8 hi = *(const bf16x8*)(q + 16);
      bfrag[ni] = __builtin_shufflevector(lo, hi, 0, 1, 2, 3, 4, 5, 6, 7,
                                                  8, 9, 10, 11, 12, 13, 14, 15);
    }
#pragma unroll
    for (int mi = 0; mi < 2; ++mi)
#pragma unroll
      for (int ni = 0; ni < 4; ++ni)
        acc[mi][ni] = __builtin_amdgcn_wmma_f32_16x16x32_bf16(
            false, afrag[mi], false, bfrag[ni],
            (short)0, acc[mi][ni], false, false);
  };

  const int nk = K / BK;
#ifdef USE_ASYNC_LDS
  load_tile(0, 0);                     // ASYNCcnt = 4 per wave
  int p = 0;
  for (int i = 0; i < nk; ++i) {
    if (i + 1 < nk) {
      load_tile(p ^ 1, (i + 1) * BK);  // ASYNCcnt = 8
      wait_async<4>();                 // tile i fully in LDS
    } else {
      wait_async<0>();
    }
    __syncthreads();                   // all waves' tile-i data visible
    compute_tile(p);                   // overlaps with tile i+1 streaming in
    if (i + 1 < nk) __syncthreads();   // done reading buf p before reuse
    p ^= 1;
  }
#else
  for (int i = 0; i < nk; ++i) {
    load_tile(0, i * BK);
    __syncthreads();
    compute_tile(0);
    __syncthreads();
  }
#endif

  // C/D layout: VGPR r -> row r (lanes 0-15) / row r+8 (lanes 16-31), col = lane&15
#pragma unroll
  for (int mi = 0; mi < 2; ++mi) {
    int mbase = block_m + wm * 32 + mi * 16 + ksel * 8;
#pragma unroll
    for (int ni = 0; ni < 4; ++ni) {
      int col = block_n + wn * 64 + ni * 16 + lhalf;
#pragma unroll
      for (int r = 0; r < 8; ++r)
        C[(size_t)(mbase + r) * N + col] = acc[mi][ni][r];
    }
  }
}

// ---------------------------------------------------------------------------
// Launch: concat, then per layer: phi_pack -> pack_w -> gemm.
// Workspace (256B-aligned offsets):
//   phi : 4096*9216*2  = 75,497,472 B
//   wt  : 9216*1024*2  = 18,874,368 B
//   xb  : 4096*1024*4  = 16,777,216 B   (safe to reuse: phi is packed first)
// ---------------------------------------------------------------------------
extern "C" void kernel_launch(void* const* d_in, const int* in_sizes, int n_in,
                              void* d_out, int out_size, void* d_ws, size_t ws_size,
                              hipStream_t stream) {
  (void)in_sizes; (void)n_in; (void)out_size; (void)ws_size;
  const float* y = (const float*)d_in[0];
  const float* u = (const float*)d_in[1];

  char*  ws  = (char*)d_ws;
  bf16*  phi = (bf16*)ws;
  bf16*  wt  = (bf16*)(ws + (size_t)75497472);
  float* xb  = (float*)(ws + (size_t)75497472 + 18874368);

  kan_concat<<<(BATCH * 512 + 255) / 256, 256, 0, stream>>>(y, u, xb);

  const int widths[5] = {512, 1024, 1024, 1024, 256};
  const float* cur = xb;
  for (int l = 0; l < 4; ++l) {
    const int din  = widths[l];
    const int dout = widths[l + 1];
    const int Kd   = din * 9;

    const float* coef = (const float*)d_in[2 + 4 * l];
    const float* sb   = (const float*)d_in[3 + 4 * l];
    const float* ss   = (const float*)d_in[4 + 4 * l];

    kan_phi_pack<<<(BATCH * din + 255) / 256, 256, 0, stream>>>(cur, din, Kd, phi);
    kan_pack_w<<<(din * dout + 255) / 256, 256, 0, stream>>>(coef, sb, ss, din, dout, Kd, wt);

    float* outp = (l == 3) ? (float*)d_out : xb;
    dim3 grid(BATCH / BM, dout / BN);
    kan_gemm<<<grid, 256, 0, stream>>>(phi, wt, outp, BATCH, dout, Kd);
    cur = outp;
  }
}
